// W4A16Linear_15023795602194
// MI455X (gfx1250) — matrix operand order repaired and checked by reading, not compiled, not run on hardware
//
#include <hip/hip_runtime.h>

// ---------------------------------------------------------------------------
// Fused NVFP4 fake-quant (per-32-block scale, nearest-LUT, ties-down) +
// bf16 WMMA GEMM:  out[M,N] = x[M,K] . dq(w)[N,K]^T     (M=128,N=11008,K=4096)
//
// Roofline: fp32 weight stream (180 MB) @ 23.3 TB/s ~ 8us floor; weight is
// read exactly once per M-half (second read hits the 192 MB L2). x is
// pre-converted to bf16 once (2 MB -> 1 MB in d_ws) so the GEMM K-loop does
// zero A-side conversion VALU and no WMMA->VALU hazard bubbles. The per-block
// absmax lane swap uses v_permlanex16_b32 (1 VALU op) instead of
// ds_bpermute_b32 + s_wait_dscnt.
// ---------------------------------------------------------------------------

typedef __attribute__((ext_vector_type(16))) __bf16        v16bf;
typedef __attribute__((ext_vector_type(8)))  float         v8f;
typedef __attribute__((ext_vector_type(4)))  unsigned int  u32x4;

__device__ __forceinline__ unsigned short f2bf_u(float f) {
    // round-to-nearest-even fp32 -> bf16 bit pattern
    unsigned u = __float_as_uint(f);
    u += 0x7FFFu + ((u >> 16) & 1u);
    return (unsigned short)(u >> 16);
}

__device__ __forceinline__ __bf16 f2bf(float f) {
    unsigned short h = f2bf_u(f);
    __bf16 r;
    __builtin_memcpy(&r, &h, 2);
    return r;
}

// lane <-> lane^16 exchange (wave32): single VALU v_permlanex16_b32
__device__ __forceinline__ float xhalf_swap(float v) {
#if __has_builtin(__builtin_amdgcn_permlanex16)
    int s = __float_as_int(v);
    int r = __builtin_amdgcn_permlanex16(s, s, 0x76543210, (int)0xFEDCBA98,
                                         true, false);
    return __int_as_float(r);
#else
    return __shfl_xor(v, 16);
#endif
}

// ---- prologue: x fp32 -> bf16 into workspace (8 elements / thread) --------
__global__ __launch_bounds__(256)
void cvt_x_bf16_kernel(const float* __restrict__ x,
                       unsigned short* __restrict__ xb, int n) {
    int i = (blockIdx.x * 256 + threadIdx.x) * 8;
    if (i >= n) return;
    float4 f0 = *reinterpret_cast<const float4*>(x + i);
    float4 f1 = *reinterpret_cast<const float4*>(x + i + 4);
    u32x4 o;
    o[0] = (unsigned)f2bf_u(f0.x) | ((unsigned)f2bf_u(f0.y) << 16);
    o[1] = (unsigned)f2bf_u(f0.z) | ((unsigned)f2bf_u(f0.w) << 16);
    o[2] = (unsigned)f2bf_u(f1.x) | ((unsigned)f2bf_u(f1.y) << 16);
    o[3] = (unsigned)f2bf_u(f1.z) | ((unsigned)f2bf_u(f1.w) << 16);
    *reinterpret_cast<u32x4*>(xb + i) = o;
}

// ---- fused dequant + GEMM -------------------------------------------------
__global__ __launch_bounds__(128)
void w4a16_nvfp4_wmma_kernel(const unsigned short* __restrict__ xb,
                             const float* __restrict__ w,
                             float* __restrict__ out,
                             int M, int N, int K) {
    const int lane  = threadIdx.x & 31;
    const int wave  = threadIdx.x >> 5;
    const int ntile = blockIdx.x * 4 + wave;   // one 16-wide N tile per wave
    const int n0    = ntile * 16;
    if (n0 >= N) return;                       // wave-uniform
    const int mbase = blockIdx.y * 64;         // 4 M-tiles per wave

    const int half = lane >> 4;                // 0: lanes 0-15, 1: lanes 16-31
    const int l15  = lane & 15;

    // B-matrix (32x16 bf16): lane l<16 holds w[n0+l][k0..k0+15],
    //                        lane l>=16 holds w[n0+l-16][k0+16..k0+31]
    const float* wrow = w + (size_t)(n0 + l15) * K + half * 16;

    // A-matrix (16x32 bf16): lane chunk1 at k0+half*16, chunk2 at k0+half*16+8
    const unsigned short* xrow0 = xb + (size_t)(mbase + 0 * 16 + l15) * K + half * 16;
    const unsigned short* xrow1 = xb + (size_t)(mbase + 1 * 16 + l15) * K + half * 16;
    const unsigned short* xrow2 = xb + (size_t)(mbase + 2 * 16 + l15) * K + half * 16;
    const unsigned short* xrow3 = xb + (size_t)(mbase + 3 * 16 + l15) * K + half * 16;

    v8f acc0 = {}, acc1 = {}, acc2 = {}, acc3 = {};

    union AV { v16bf v; u32x4 u[2]; };

    for (int k0 = 0; k0 < K; k0 += 32) {
        // ---- stream one 32-K slab of the weight row (fp32, read once) ----
        const float4* wp = reinterpret_cast<const float4*>(wrow + k0);
        float4 q0 = wp[0], q1 = wp[1], q2 = wp[2], q3 = wp[3];
        __builtin_prefetch(wrow + k0 + 64, 0, 1);   // global_prefetch_b8
        float bvals[16] = { q0.x, q0.y, q0.z, q0.w,
                            q1.x, q1.y, q1.z, q1.w,
                            q2.x, q2.y, q2.z, q2.w,
                            q3.x, q3.y, q3.z, q3.w };

        // ---- A tiles: straight b128 bf16 loads, no conversion VALU -------
        AV A0, A1, A2, A3;
        A0.u[0] = *reinterpret_cast<const u32x4*>(xrow0 + k0);
        A0.u[1] = *reinterpret_cast<const u32x4*>(xrow0 + k0 + 8);
        A1.u[0] = *reinterpret_cast<const u32x4*>(xrow1 + k0);
        A1.u[1] = *reinterpret_cast<const u32x4*>(xrow1 + k0 + 8);
        A2.u[0] = *reinterpret_cast<const u32x4*>(xrow2 + k0);
        A2.u[1] = *reinterpret_cast<const u32x4*>(xrow2 + k0 + 8);
        A3.u[0] = *reinterpret_cast<const u32x4*>(xrow3 + k0);
        A3.u[1] = *reinterpret_cast<const u32x4*>(xrow3 + k0 + 8);

        // ---- per-32-block absmax: (lane, lane^16) pair spans the block ----
        float amax = 0.f;
#pragma unroll
        for (int i = 0; i < 16; ++i) amax = fmaxf(amax, fabsf(bvals[i]));
        amax = fmaxf(amax, xhalf_swap(amax));
        const float scale = fmaxf(amax * (1.0f / 6.0f), 1e-12f);

        // NVFP4 LUT levels/mid-points pre-multiplied by scale (no division;
        // `a > mid` keeps the reference's ties-round-down behavior).
        const float s05 = 0.5f  * scale, s10 = scale,        s15 = 1.5f * scale;
        const float s20 = 2.0f  * scale, s30 = 3.0f * scale, s40 = 4.0f * scale;
        const float s60 = 6.0f  * scale;
        const float t02 = 0.25f * scale, t07 = 0.75f * scale, t12 = 1.25f * scale;
        const float t17 = 1.75f * scale, t25 = 2.5f  * scale, t35 = 3.5f  * scale;
        const float t50 = 5.0f  * scale;

        v16bf B;
#pragma unroll
        for (int i = 0; i < 16; ++i) {
            const float v = bvals[i];
            const float a = fabsf(v);
            float q = a > t50 ? s60
                    : a > t35 ? s40
                    : a > t25 ? s30
                    : a > t17 ? s20
                    : a > t12 ? s15
                    : a > t07 ? s10
                    : a > t02 ? s05 : 0.0f;
            B[i] = f2bf(copysignf(q, v));
        }

        acc0 = __builtin_amdgcn_wmma_f32_16x16x32_bf16(false, A0.v, false, B, (short)0, acc0, false, false);
        acc1 = __builtin_amdgcn_wmma_f32_16x16x32_bf16(false, A1.v, false, B, (short)0, acc1, false, false);
        acc2 = __builtin_amdgcn_wmma_f32_16x16x32_bf16(false, A2.v, false, B, (short)0, acc2, false, false);
        acc3 = __builtin_amdgcn_wmma_f32_16x16x32_bf16(false, A3.v, false, B, (short)0, acc3, false, false);
    }

    // ---- store: C/D layout VGPR j -> M = j + half*8, lane -> N = l15 ----
    const int ncol = n0 + l15;
#pragma unroll
    for (int j = 0; j < 8; ++j) {
        out[(size_t)(mbase +  0 + j + half * 8) * N + ncol] = acc0[j];
        out[(size_t)(mbase + 16 + j + half * 8) * N + ncol] = acc1[j];
        out[(size_t)(mbase + 32 + j + half * 8) * N + ncol] = acc2[j];
        out[(size_t)(mbase + 48 + j + half * 8) * N + ncol] = acc3[j];
    }
}

extern "C" void kernel_launch(void* const* d_in, const int* in_sizes, int n_in,
                              void* d_out, int out_size, void* d_ws, size_t ws_size,
                              hipStream_t stream) {
    const float* x = (const float*)d_in[0];   // [1,128,4096] fp32
    const float* w = (const float*)d_in[1];   // [11008,4096] fp32
    float* out = (float*)d_out;               // [1,128,11008] fp32

    const int K = 4096;
    const int M = in_sizes[0] / K;            // 128
    const int N = in_sizes[1] / K;            // 11008
    const int nx = in_sizes[0];               // 524288

    unsigned short* xb = (unsigned short*)d_ws;   // 1 MB bf16 copy of x

    // 1) x fp32 -> bf16 (RNE) into workspace
    cvt_x_bf16_kernel<<<(nx / 8 + 255) / 256, 256, 0, stream>>>(x, xb, nx);

    // 2) fused dequant + WMMA GEMM
    dim3 grid((N / 16 + 3) / 4, M / 64);      // (172, 2): 4 waves/block, 16 N-cols/wave, 64 M-rows
    dim3 block(128);
    w4a16_nvfp4_wmma_kernel<<<grid, block, 0, stream>>>(xb, w, out, M, N, K);
}
